// MambaDecoderLayer_81484119540440
// MI455X (gfx1250) — matrix operand order repaired and checked by reading, not compiled, not run on hardware
//
#include <hip/hip_runtime.h>
#include <hip/hip_bf16.h>

// ---------------------------------------------------------------------------
// Mamba decoder layer for gfx1250 (MI455X):
//  - f16 WMMA GEMMs (v_wmma_f32_16x16x32_f16), 128x128 block tile,
//    double-buffered LDS staged via async global->LDS (ASYNCcnt) when available
//  - chunked (64-way time-parallel) selective scan
// ---------------------------------------------------------------------------

typedef __attribute__((ext_vector_type(16))) _Float16 v16h;
typedef __attribute__((ext_vector_type(8)))  _Float16 v8h;
typedef __attribute__((ext_vector_type(8)))  float    v8f;
typedef __attribute__((ext_vector_type(4)))  int      v4i;

constexpr int S_    = 2048;          // sequence length
constexpr int H_    = 2048;          // hidden size
constexpr int I_    = 4096;          // intermediate size
constexpr int R_    = 128;           // time-step rank
constexpr int E_    = R_ + 32;       // 160 = x_proj output (dt|B|C)
constexpr int TWOI_ = 2 * I_;        // 8192

constexpr int TK  = 64;              // K depth per LDS stage
constexpr int LDA = 72;              // padded LDS row stride (halves)

constexpr int CL_ = 32;              // scan chunk length
constexpr int CH_ = S_ / CL_;        // 64 chunks

#if defined(__AMDGCN__) && __has_builtin(__builtin_amdgcn_global_load_async_to_lds_b128)
#define ASYNC_LDS 1
#else
#define ASYNC_LDS 0
#endif

#define GLOBAL_AS __attribute__((address_space(1)))
#define LDS_AS    __attribute__((address_space(3)))

// 16B global -> LDS copy: async DMA path on CDNA5, VGPR round-trip otherwise.
__device__ __forceinline__ void cp_b128(const _Float16* __restrict__ g,
                                        _Float16* l) {
#if ASYNC_LDS
  __builtin_amdgcn_global_load_async_to_lds_b128(
      (GLOBAL_AS v4i*)(void*)g, (LDS_AS v4i*)(void*)l, 0, 0);
#else
  *(v8h*)l = *(const v8h*)g;
#endif
}

__device__ __forceinline__ void wait_stage() {
#if ASYNC_LDS
#if __has_builtin(__builtin_amdgcn_s_wait_asynccnt)
  __builtin_amdgcn_s_wait_asynccnt(0);
#else
  asm volatile("s_wait_asynccnt 0" ::: "memory");
#endif
#endif
  __syncthreads();
}

// ---------------------------------------------------------------------------
// f32 -> f16 cast
// ---------------------------------------------------------------------------
__global__ void k_cast_f16(const float* __restrict__ in, _Float16* __restrict__ out, int n) {
  int i = (blockIdx.x * 256 + threadIdx.x) * 4;
#pragma unroll
  for (int j = 0; j < 4; ++j)
    if (i + j < n) out[i + j] = (_Float16)in[i + j];
}

// ---------------------------------------------------------------------------
// fused residual-add + RMSNorm: residual (f32 output #2) + h (f16 GEMM operand)
// ---------------------------------------------------------------------------
__global__ __launch_bounds__(256) void k_add_rmsnorm(
    const float* __restrict__ hs, const float* __restrict__ res,
    const float* __restrict__ w, float* __restrict__ res_out,
    _Float16* __restrict__ h16) {
  const int row = blockIdx.x, tid = threadIdx.x;
  const size_t base = (size_t)row * H_;
  float v[8];
  float ss = 0.f;
#pragma unroll
  for (int j = 0; j < 8; ++j) {
    int c = tid * 8 + j;
    float x = hs[base + c] + res[base + c];
    v[j] = x;
    ss += x * x;
  }
  __shared__ float red[256];
  red[tid] = ss;
  __syncthreads();
  for (int s = 128; s > 0; s >>= 1) {
    if (tid < s) red[tid] += red[tid + s];
    __syncthreads();
  }
  const float rs = rsqrtf(red[0] / (float)H_ + 1e-5f);
#pragma unroll
  for (int j = 0; j < 8; ++j) {
    int c = tid * 8 + j;
    res_out[base + c] = v[j];
    h16[base + c] = (_Float16)(v[j] * rs * w[c]);
  }
}

// ---------------------------------------------------------------------------
// Tiled WMMA GEMM:  D[M,Nout] = A[M,K](f16) * W[Nout,K]^T(f16), f32 accum.
// 256 threads (8 waves as 4Mx2N), block tile 128x128, wave tile 32Mx64N.
// Double-buffered LDS stages, filled by async global->LDS when available.
// EPI 0: store f32.  EPI 1: softplus(acc + bias[col]).
// ---------------------------------------------------------------------------
__device__ __forceinline__ void stage_tiles(
    const _Float16* __restrict__ A, const _Float16* __restrict__ Bw,
    _Float16* sAb, _Float16* sBb, int m0, int n0, int kt, int Kd, int tid) {
#pragma unroll
  for (int i = 0; i < 4; ++i) {
    int c = tid + i * 256;
    int row = c >> 3, kc = (c & 7) * 8;
    cp_b128(A + (size_t)(m0 + row) * Kd + kt + kc, &sAb[row * LDA + kc]);
    cp_b128(Bw + (size_t)(n0 + row) * Kd + kt + kc, &sBb[row * LDA + kc]);
  }
}

template <int EPI>
__global__ __launch_bounds__(256) void k_wmma_gemm(
    const _Float16* __restrict__ A, const _Float16* __restrict__ Bw,
    float* __restrict__ Dst, const float* __restrict__ bias,
    int M, int Nout, int Kd) {
  __shared__ __align__(16) _Float16 sA[2][128 * LDA];
  __shared__ __align__(16) _Float16 sB[2][128 * LDA];

  const int tid  = threadIdx.x;
  const int lane = tid & 31;
  const int wave = tid >> 5;
  const int wm = wave >> 1, wn = wave & 1;   // 4 x 2 wave grid
  const int g  = lane >> 4, ml = lane & 15;
  const int m0 = blockIdx.y * 128;
  const int n0 = blockIdx.x * 128;

  v8f acc[2][4] = {};

  stage_tiles(A, Bw, sA[0], sB[0], m0, n0, 0, Kd, tid);
  wait_stage();
  int cur = 0;

  for (int k0 = 0; k0 < Kd; k0 += TK) {
    if (k0 + TK < Kd)
      stage_tiles(A, Bw, sA[cur ^ 1], sB[cur ^ 1], m0, n0, k0 + TK, Kd, tid);

#pragma unroll
    for (int kk = 0; kk < TK; kk += 32) {
      v16h af[2], bf[4];
      // A fragment (ISA 16-bit A layout): lanes 0-15 -> K [kk..+7],[kk+16..+23]
#pragma unroll
      for (int tm = 0; tm < 2; ++tm) {
        int base = (wm * 32 + tm * 16 + ml) * LDA + kk + (g ? 8 : 0);
        v8h lo = *(const v8h*)&sA[cur][base];
        v8h hi = *(const v8h*)&sA[cur][base + 16];
        af[tm] = __builtin_shufflevector(lo, hi, 0, 1, 2, 3, 4, 5, 6, 7, 8, 9,
                                         10, 11, 12, 13, 14, 15);
      }
      // B fragment (same as A): lanes 0-15 -> K [kk..+7],[kk+16..+23]
#pragma unroll
      for (int tn = 0; tn < 4; ++tn) {
        int base = (wn * 64 + tn * 16 + ml) * LDA + kk + (g ? 8 : 0);
        v8h lo = *(const v8h*)&sB[cur][base];
        v8h hi = *(const v8h*)&sB[cur][base + 16];
        bf[tn] = __builtin_shufflevector(lo, hi, 0, 1, 2, 3, 4, 5, 6, 7, 8, 9,
                                         10, 11, 12, 13, 14, 15);
      }
#pragma unroll
      for (int tm = 0; tm < 2; ++tm)
#pragma unroll
        for (int tn = 0; tn < 4; ++tn)
          acc[tm][tn] = __builtin_amdgcn_wmma_f32_16x16x32_f16(
              false, af[tm], false, bf[tn], (short)0, acc[tm][tn], false, false);
    }
    wait_stage();
    cur ^= 1;
  }

  // epilogue: C/D layout = 8 VGPRs; lanes 0-15 M=r, lanes 16-31 M=8+r
#pragma unroll
  for (int tm = 0; tm < 2; ++tm)
#pragma unroll
    for (int tn = 0; tn < 4; ++tn) {
      int col = n0 + wn * 64 + tn * 16 + ml;
      if (col < Nout) {
        int rbase = m0 + wm * 32 + tm * 16 + (g ? 8 : 0);
#pragma unroll
        for (int r = 0; r < 8; ++r) {
          float v = acc[tm][tn][r];
          if (EPI == 1) {
            v += bias[col];
            v = (v > 20.f) ? v : log1pf(__expf(v));
          }
          Dst[(size_t)(rbase + r) * Nout + col] = v;
        }
      }
    }
}

// ---------------------------------------------------------------------------
// depthwise causal conv (K=4) + SiLU;  x = first I columns of xz [S, 2I]
// ---------------------------------------------------------------------------
__global__ void k_conv_silu(const float* __restrict__ xz,
                            const float* __restrict__ cw,
                            const float* __restrict__ cb,
                            float* __restrict__ xc, _Float16* __restrict__ xc16) {
  int idx = blockIdx.x * 256 + threadIdx.x;
  if (idx >= S_ * I_) return;
  int t = idx / I_, i = idx % I_;
  float acc = cb[i];
#pragma unroll
  for (int k = 0; k < 4; ++k) {
    int tt = t - 3 + k;
    if (tt >= 0) acc += cw[i * 4 + k] * xz[(size_t)tt * TWOI_ + i];
  }
  float s = acc / (1.f + __expf(-acc));   // SiLU
  xc[idx] = s;
  xc16[idx] = (_Float16)s;
}

// ---------------------------------------------------------------------------
// slice dt columns out of dbc [S,160] -> f16 [S,128]
// ---------------------------------------------------------------------------
__global__ void k_extract_dt(const float* __restrict__ dbc,
                             _Float16* __restrict__ dtlow) {
  int idx = blockIdx.x * 256 + threadIdx.x;
  if (idx >= S_ * R_) return;
  int s = idx / R_, r = idx % R_;
  dtlow[idx] = (_Float16)dbc[(size_t)s * E_ + r];
}

// ---------------------------------------------------------------------------
// chunked selective scan, pass A: per (channel, chunk) local scan from h=0.
// Emits P = prod(a) and Hl = local end state (16 each).
// ---------------------------------------------------------------------------
__global__ __launch_bounds__(256) void k_scan_partial(
    const float* __restrict__ dbc, const float* __restrict__ dt,
    const float* __restrict__ xc, const float* __restrict__ A_log,
    float* __restrict__ P, float* __restrict__ Hl) {
  const int tid = threadIdx.x;
  const int ch = blockIdx.x * 256 + tid;
  const int chunk = blockIdx.y;
  float Arow[16], p[16], h[16];
#pragma unroll
  for (int n = 0; n < 16; ++n) {
    Arow[n] = -__expf(A_log[(size_t)ch * 16 + n]);
    p[n] = 1.f;
    h[n] = 0.f;
  }
  __shared__ float sBC[CL_][32];
#pragma unroll
  for (int j = 0; j < 4; ++j) {
    int idx = tid + j * 256;
    int tt = idx >> 5, c = idx & 31;
    sBC[tt][c] = dbc[(size_t)(chunk * CL_ + tt) * E_ + 128 + c];
  }
  __syncthreads();
  for (int tt = 0; tt < CL_; ++tt) {
    int t = chunk * CL_ + tt;
    float sdt = dt[(size_t)t * I_ + ch];
    float dtx = sdt * xc[(size_t)t * I_ + ch];
#pragma unroll
    for (int n = 0; n < 16; ++n) {
      float a = __expf(sdt * Arow[n]);
      h[n] = a * h[n] + dtx * sBC[tt][n];
      p[n] *= a;
    }
  }
  size_t base = ((size_t)chunk * I_ + ch) * 16;
#pragma unroll
  for (int n = 0; n < 16; ++n) {
    P[base + n] = p[n];
    Hl[base + n] = h[n];
  }
}

// ---------------------------------------------------------------------------
// pass B: serial carry over 64 chunks per channel (cheap), emits per-chunk
// initial state S0.
// ---------------------------------------------------------------------------
__global__ __launch_bounds__(256) void k_scan_carry(
    const float* __restrict__ P, const float* __restrict__ Hl,
    float* __restrict__ S0) {
  const int ch = blockIdx.x * 256 + threadIdx.x;
  float carry[16];
#pragma unroll
  for (int n = 0; n < 16; ++n) carry[n] = 0.f;
  for (int c = 0; c < CH_; ++c) {
    size_t base = ((size_t)c * I_ + ch) * 16;
#pragma unroll
    for (int n = 0; n < 16; ++n) S0[base + n] = carry[n];
#pragma unroll
    for (int n = 0; n < 16; ++n)
      carry[n] = P[base + n] * carry[n] + Hl[base + n];
  }
}

// ---------------------------------------------------------------------------
// pass C: recompute each chunk from its true initial state; emit y.
// ---------------------------------------------------------------------------
__global__ __launch_bounds__(256) void k_scan_final(
    const float* __restrict__ dbc, const float* __restrict__ dt,
    const float* __restrict__ xc, const float* __restrict__ A_log,
    const float* __restrict__ S0, float* __restrict__ y) {
  const int tid = threadIdx.x;
  const int ch = blockIdx.x * 256 + tid;
  const int chunk = blockIdx.y;
  float Arow[16], h[16];
  size_t sbase = ((size_t)chunk * I_ + ch) * 16;
#pragma unroll
  for (int n = 0; n < 16; ++n) {
    Arow[n] = -__expf(A_log[(size_t)ch * 16 + n]);
    h[n] = S0[sbase + n];
  }
  __shared__ float sBC[CL_][32];
#pragma unroll
  for (int j = 0; j < 4; ++j) {
    int idx = tid + j * 256;
    int tt = idx >> 5, c = idx & 31;
    sBC[tt][c] = dbc[(size_t)(chunk * CL_ + tt) * E_ + 128 + c];
  }
  __syncthreads();
  for (int tt = 0; tt < CL_; ++tt) {
    int t = chunk * CL_ + tt;
    float sdt = dt[(size_t)t * I_ + ch];
    float dtx = sdt * xc[(size_t)t * I_ + ch];
    float acc = 0.f;
#pragma unroll
    for (int n = 0; n < 16; ++n) {
      h[n] = __expf(sdt * Arow[n]) * h[n] + dtx * sBC[tt][n];
      acc += h[n] * sBC[tt][16 + n];
    }
    y[(size_t)t * I_ + ch] = acc;
  }
}

// ---------------------------------------------------------------------------
// y = (y + D*x) * silu(z)  -> f16 operand of out_proj
// ---------------------------------------------------------------------------
__global__ void k_gate(const float* __restrict__ y, const float* __restrict__ xc,
                       const float* __restrict__ xz, const float* __restrict__ Dp,
                       _Float16* __restrict__ yg16) {
  int idx = blockIdx.x * 256 + threadIdx.x;
  if (idx >= S_ * I_) return;
  int t = idx / I_, i = idx % I_;
  float zv = xz[(size_t)t * TWOI_ + I_ + i];
  float gate = zv / (1.f + __expf(-zv));
  yg16[idx] = (_Float16)((y[idx] + Dp[i] * xc[idx]) * gate);
}

// ---------------------------------------------------------------------------
extern "C" void kernel_launch(void* const* d_in, const int* in_sizes, int n_in,
                              void* d_out, int out_size, void* d_ws, size_t ws_size,
                              hipStream_t stream) {
  const float* hs     = (const float*)d_in[0];
  const float* res    = (const float*)d_in[1];
  const float* norm_w = (const float*)d_in[2];
  const float* W_in   = (const float*)d_in[3];
  const float* conv_w = (const float*)d_in[4];
  const float* conv_b = (const float*)d_in[5];
  const float* W_x    = (const float*)d_in[6];
  const float* W_dt   = (const float*)d_in[7];
  const float* b_dt   = (const float*)d_in[8];
  const float* A_log  = (const float*)d_in[9];
  const float* Dp     = (const float*)d_in[10];
  const float* W_out  = (const float*)d_in[11];

  float* out     = (float*)d_out;
  float* res_out = out + (size_t)S_ * H_;

  char* wsp = (char*)d_ws;
  auto take = [&](size_t bytes) -> void* {
    void* p = wsp;
    wsp += (bytes + 255) & ~(size_t)255;
    return p;
  };
  _Float16* hf16    = (_Float16*)take((size_t)S_ * H_ * 2);
  _Float16* Win16   = (_Float16*)take((size_t)TWOI_ * H_ * 2);
  _Float16* Wx16    = (_Float16*)take((size_t)E_ * I_ * 2);
  _Float16* Wdt16   = (_Float16*)take((size_t)I_ * R_ * 2);
  _Float16* Wout16  = (_Float16*)take((size_t)H_ * I_ * 2);
  float*    xz      = (float*)   take((size_t)S_ * TWOI_ * 4);
  float*    xc      = (float*)   take((size_t)S_ * I_ * 4);
  _Float16* xc16    = (_Float16*)take((size_t)S_ * I_ * 2);
  float*    dbc     = (float*)   take((size_t)S_ * E_ * 4);
  _Float16* dtlow16 = (_Float16*)take((size_t)S_ * R_ * 2);
  float*    dtbuf   = (float*)   take((size_t)S_ * I_ * 4);
  float*    ybuf    = (float*)   take((size_t)S_ * I_ * 4);
  _Float16* yg16    = (_Float16*)take((size_t)S_ * I_ * 2);
  float*    Pbuf    = (float*)   take((size_t)CH_ * I_ * 16 * 4);
  float*    Hbuf    = (float*)   take((size_t)CH_ * I_ * 16 * 4);
  float*    S0buf   = (float*)   take((size_t)CH_ * I_ * 16 * 4);

  auto cast = [&](const float* src, _Float16* dst, int n) {
    k_cast_f16<<<(n + 1023) / 1024, 256, 0, stream>>>(src, dst, n);
  };
  cast(W_in,  Win16,  TWOI_ * H_);
  cast(W_x,   Wx16,   E_ * I_);
  cast(W_dt,  Wdt16,  I_ * R_);
  cast(W_out, Wout16, H_ * I_);

  // 1) fused add + RMSNorm
  k_add_rmsnorm<<<S_, 256, 0, stream>>>(hs, res, norm_w, res_out, hf16);

  // 2) in_proj: [S,H] x [H,2I] -> xz
  k_wmma_gemm<0><<<dim3(TWOI_ / 128, S_ / 128), 256, 0, stream>>>(
      hf16, Win16, xz, nullptr, S_, TWOI_, H_);

  // 3) depthwise causal conv + SiLU
  k_conv_silu<<<(S_ * I_) / 256, 256, 0, stream>>>(xz, conv_w, conv_b, xc, xc16);

  // 4) x_proj: [S,I] x [I,160] -> dbc
  k_wmma_gemm<0><<<dim3((E_ + 127) / 128, S_ / 128), 256, 0, stream>>>(
      xc16, Wx16, dbc, nullptr, S_, E_, I_);

  // 5) dt_proj: slice dt, [S,128] x [128,I], fused softplus(+bias)
  k_extract_dt<<<(S_ * R_) / 256, 256, 0, stream>>>(dbc, dtlow16);
  k_wmma_gemm<1><<<dim3(I_ / 128, S_ / 128), 256, 0, stream>>>(
      dtlow16, Wdt16, dtbuf, b_dt, S_, I_, R_);

  // 6) chunked selective scan: partial -> carry -> final
  k_scan_partial<<<dim3(I_ / 256, CH_), 256, 0, stream>>>(
      dbc, dtbuf, xc, A_log, Pbuf, Hbuf);
  k_scan_carry<<<I_ / 256, 256, 0, stream>>>(Pbuf, Hbuf, S0buf);
  k_scan_final<<<dim3(I_ / 256, CH_), 256, 0, stream>>>(
      dbc, dtbuf, xc, A_log, S0buf, ybuf);

  // 7) gate: (y + D*x) * silu(z)
  k_gate<<<(S_ * I_) / 256, 256, 0, stream>>>(ybuf, xc, xz, Dp, yg16);

  // 8) out_proj: [S,I] x [I,H] -> out
  k_wmma_gemm<0><<<dim3(H_ / 128, S_ / 128), 256, 0, stream>>>(
      yg16, Wout16, out, nullptr, S_, H_, I_);
}
